// HGTEncoder_33208687133239
// MI455X (gfx1250) — matrix operand order repaired and checked by reading, not compiled, not run on hardware
//
#include <hip/hip_runtime.h>
#include <cstdint>
#include <cstddef>

// ---------------------------------------------------------------------------
// HGT encoder for gfx1250 (MI455X).
//  - All dense GEMMs: bf16 operands (pre-converted in global), f32 accumulate
//    via v_wmma_f32_16x16x32_bf16. Tiles staged with the CDNA5 async DMA path
//    (global_load_async_to_lds_b128 + s_wait_asynccnt), double-buffered LDS.
//  - Edge phase: L2-resident wave-per-edge scoring + 3-pass atomic softmax.
// ---------------------------------------------------------------------------

#define BM 64
#define BN 64
#define BK 32
#define LDSP (BK + 8)   // ushort row stride: 80B, avoids DS bank conflicts

typedef __bf16 bf16_t;
typedef bf16_t v16bf __attribute__((ext_vector_type(16)));
typedef float  v8f   __attribute__((ext_vector_type(8)));
typedef unsigned short ushort_t;

union Frag32B { uint4 u4[2]; v16bf v; };

__device__ __forceinline__ ushort_t f2bf(float f) {
  unsigned int u = __float_as_uint(f);
  u += 0x7FFFu + ((u >> 16) & 1u);   // round-to-nearest-even
  return (ushort_t)(u >> 16);
}

// ---------------------------------------------------------------------------
// GEMM: C[M,N] = A[M,K] @ W[N,K]^T (+ bias), A/W bf16, C f32 (+ optional bf16
// copy C2 for downstream GEMM reuse). K % 32 == 0, N % 64 == 0.
// 128 threads = 4 waves; each wave owns a 32x32 quadrant (2x2 WMMA tiles).
// ---------------------------------------------------------------------------
__global__ __launch_bounds__(128)
void gemm_bf16_kernel(const ushort_t* __restrict__ A, int M, int K, int lda,
                      const ushort_t* __restrict__ W,     // [N,K]
                      const float* __restrict__ bias,     // may be null
                      float* __restrict__ C, int N, int ldc,
                      ushort_t* __restrict__ C2, int ldc2) // may be null
{
  __shared__ ushort_t As[2][BM][LDSP];
  __shared__ ushort_t Bs[2][BN][LDSP];

  const int m0   = blockIdx.x * BM;
  const int n0   = blockIdx.y * BN;
  const int tid  = threadIdx.x;
  const int wave = tid >> 5;
  const int lane = tid & 31;
  const int wm   = (wave & 1) * 32;
  const int wn   = (wave >> 1) * 32;

  v8f acc[2][2] = {};

  // staging assignment: 2 threads per tile row; each thread copies 32B (2x b128)
  const int srow = tid >> 1;             // 0..63
  const int scol = (tid & 1) * 16;       // ushort offset 0 or 16
  const int arow = (m0 + srow < M) ? (m0 + srow) : (M - 1);  // clamp; junk rows unread
  const int brow = n0 + srow;

  // async DMA one 64x32 bf16 tile pair into LDS buffer `buf` for k-offset k0
  auto stage = [&](int buf, int k0) {
    unsigned a_off = ((unsigned)arow * (unsigned)lda + (unsigned)(k0 + scol)) * 2u;
    unsigned b_off = ((unsigned)brow * (unsigned)K   + (unsigned)(k0 + scol)) * 2u;
    unsigned a_lds = (unsigned)(uintptr_t)&As[buf][srow][scol];
    unsigned b_lds = (unsigned)(uintptr_t)&Bs[buf][srow][scol];
    asm volatile("global_load_async_to_lds_b128 %0, %1, %2"
                 :: "v"(a_lds), "v"(a_off), "s"(A) : "memory");
    asm volatile("global_load_async_to_lds_b128 %0, %1, %2 offset:16"
                 :: "v"(a_lds), "v"(a_off), "s"(A) : "memory");
    asm volatile("global_load_async_to_lds_b128 %0, %1, %2"
                 :: "v"(b_lds), "v"(b_off), "s"(W) : "memory");
    asm volatile("global_load_async_to_lds_b128 %0, %1, %2 offset:16"
                 :: "v"(b_lds), "v"(b_off), "s"(W) : "memory");
  };

  stage(0, 0);
  int buf = 0;
  for (int k0 = 0; k0 < K; k0 += BK) {
    if (k0 + BK < K) {
      stage(buf ^ 1, k0 + BK);                       // prefetch next tile
      asm volatile("s_wait_asynccnt 4" ::: "memory"); // current tile landed
    } else {
      asm volatile("s_wait_asynccnt 0" ::: "memory");
    }
    __syncthreads();

    // WMMA fragments from LDS (ISA 7.12.2 layouts)
    Frag32B a[2], b[2];
    const int akb = (lane >= 16) ? 8 : 0;
    #pragma unroll
    for (int i = 0; i < 2; ++i) {
      const int r = wm + i * 16 + (lane & 15);
      a[i].u4[0] = *(const uint4*)&As[buf][r][akb];
      a[i].u4[1] = *(const uint4*)&As[buf][r][akb + 16];
    }
    const int bkb = (lane < 16) ? 0 : 8;
    #pragma unroll
    for (int j = 0; j < 2; ++j) {
      const int cn = wn + j * 16 + (lane & 15);
      b[j].u4[0] = *(const uint4*)&Bs[buf][cn][bkb];
      b[j].u4[1] = *(const uint4*)&Bs[buf][cn][bkb + 16];
    }

    #pragma unroll
    for (int i = 0; i < 2; ++i)
      #pragma unroll
      for (int j = 0; j < 2; ++j)
        acc[i][j] = __builtin_amdgcn_wmma_f32_16x16x32_bf16(
            false, a[i].v, false, b[j].v, (short)0, acc[i][j], false, false);

    __syncthreads();   // everyone done reading buf before it is restaged
    buf ^= 1;
  }

  // epilogue: bias + f32 store (+ optional bf16 mirror)
  const int rhalf = (lane >= 16) ? 8 : 0;
  const int ccol  = lane & 15;
  #pragma unroll
  for (int i = 0; i < 2; ++i) {
    #pragma unroll
    for (int j = 0; j < 2; ++j) {
      const int col = n0 + wn + j * 16 + ccol;
      const float bv = bias ? bias[col] : 0.0f;
      #pragma unroll
      for (int r = 0; r < 8; ++r) {
        const int row = m0 + wm + i * 16 + rhalf + r;
        if (row < M) {
          float val = acc[i][j][r] + bv;
          C[(size_t)row * ldc + col] = val;
          if (C2) C2[(size_t)row * ldc2 + col] = f2bf(val);
        }
      }
    }
  }
}

// f32 -> bf16 elementwise
__global__ void convert_bf16_kernel(const float* __restrict__ in,
                                    ushort_t* __restrict__ out, long long n) {
  long long i = (long long)blockIdx.x * blockDim.x + threadIdx.x;
  if (i < n) out[i] = f2bf(in[i]);
}

// 128x128 block-diagonal bf16 weight from rel[4][32][32] (h,d,f):
// W[h*32+f][h*32+d] = rel[h][d][f]  so C = A @ W^T applies per-head d->f.
__global__ void build_blockdiag_kernel(const float* __restrict__ rel,
                                       ushort_t* __restrict__ Wd)
{
  int idx = blockIdx.x * blockDim.x + threadIdx.x;
  if (idx >= 128 * 128) return;
  int row = idx >> 7, col = idx & 127;
  int h = row >> 5, f = row & 31;
  int hc = col >> 5, d = col & 31;
  Wd[idx] = (h == hc) ? f2bf(rel[(h * 32 + d) * 32 + f]) : (ushort_t)0;
}

__global__ void fill_kernel(float* __restrict__ p, long long n, float v) {
  long long i = (long long)blockIdx.x * blockDim.x + threadIdx.x;
  if (i < n) p[i] = v;
}

__device__ __forceinline__ void atomic_max_float(float* addr, float v) {
  if (v >= 0.0f) atomicMax((int*)addr, __float_as_int(v));
  else           atomicMin((unsigned int*)addr, __float_as_uint(v));
}

// wave-per-edge score + scatter-max. kqv_dst layout [N,384], Q at cols 128..255.
__global__ __launch_bounds__(256)
void edge_score_kernel(const int* __restrict__ edge, int E,
                       const float* __restrict__ kqv_dst,
                       const float* __restrict__ kt,
                       const float* __restrict__ prel,
                       float* __restrict__ score, float* __restrict__ mx)
{
  int e    = (blockIdx.x * blockDim.x + threadIdx.x) >> 5;
  int lane = threadIdx.x & 31;
  if (e >= E) return;
  int si = edge[e], di = edge[E + e];
  const float* q = kqv_dst + (size_t)di * 384 + 128;
  const float* k = kt + (size_t)si * 128;
  #pragma unroll
  for (int h = 0; h < 4; ++h) {
    float p = q[h * 32 + lane] * k[h * 32 + lane];
    #pragma unroll
    for (int o = 16; o > 0; o >>= 1) p += __shfl_xor(p, o);
    if (lane == 0) {
      float s = p * prel[h] * 0.17677669529663687f;   // 1/sqrt(32)
      score[(size_t)e * 4 + h] = s;
      atomic_max_float(&mx[(size_t)di * 4 + h], s);
    }
  }
}

__global__ void edge_expsum_kernel(const int* __restrict__ edge, int E,
                                   float* __restrict__ score,
                                   const float* __restrict__ mx,
                                   float* __restrict__ den)
{
  int idx = blockIdx.x * blockDim.x + threadIdx.x;
  if (idx >= E * 4) return;
  int e = idx >> 2, h = idx & 3;
  int di = edge[E + e];
  float ex = __expf(score[idx] - mx[(size_t)di * 4 + h]);
  score[idx] = ex;
  atomicAdd(&den[(size_t)di * 4 + h], ex);
}

__global__ void edge_scatter_kernel(const int* __restrict__ edge, int E,
                                    const float* __restrict__ score,
                                    const float* __restrict__ den,
                                    const float* __restrict__ vt,
                                    float* __restrict__ agg)
{
  long long idx = (long long)blockIdx.x * blockDim.x + threadIdx.x;
  if (idx >= (long long)E * 128) return;
  int e = (int)(idx >> 7), f = (int)(idx & 127), h = f >> 5;
  int si = edge[e], di = edge[E + e];
  float alpha = score[(size_t)e * 4 + h] / den[(size_t)di * 4 + h];
  atomicAdd(&agg[(size_t)di * 128 + f], alpha * vt[(size_t)si * 128 + f]);
}

// gelu(x) with bf16 output (feeds the next GEMM's A operand)
__global__ void gelu_bf_kernel(const float* __restrict__ in,
                               ushort_t* __restrict__ out, long long n) {
  long long i = (long long)blockIdx.x * blockDim.x + threadIdx.x;
  if (i >= n) return;
  float x = in[i];
  out[i] = f2bf(0.5f * x * (1.0f + erff(x * 0.70710678118654752f)));
}

// h = sigmoid(skip) * a + (1 - sigmoid(skip)) * h
__global__ void blend_kernel(float* __restrict__ h, const float* __restrict__ a,
                             long long n, const float* __restrict__ skip, int sidx)
{
  long long i = (long long)blockIdx.x * blockDim.x + threadIdx.x;
  if (i >= n) return;
  float sk = 1.0f / (1.0f + __expf(-skip[sidx]));
  h[i] = sk * a[i] + (1.0f - sk) * h[i];
}

// wave-per-node LayerNorm(128) + exact GELU; writes f32 in place + bf16 copy
__global__ __launch_bounds__(256)
void ln_gelu_kernel(float* __restrict__ h, ushort_t* __restrict__ hbf, int N,
                    const float* __restrict__ g, const float* __restrict__ b)
{
  int w    = (blockIdx.x * blockDim.x + threadIdx.x) >> 5;
  int lane = threadIdx.x & 31;
  if (w >= N) return;
  float* x = h + (size_t)w * 128;
  ushort_t* xb = hbf + (size_t)w * 128;
  float v[4];
  #pragma unroll
  for (int i = 0; i < 4; ++i) v[i] = x[lane + 32 * i];
  float s = v[0] + v[1] + v[2] + v[3];
  #pragma unroll
  for (int o = 16; o > 0; o >>= 1) s += __shfl_xor(s, o);
  float mu = s * (1.0f / 128.0f);
  float ss = 0.0f;
  #pragma unroll
  for (int i = 0; i < 4; ++i) { float d = v[i] - mu; ss += d * d; }
  #pragma unroll
  for (int o = 16; o > 0; o >>= 1) ss += __shfl_xor(ss, o);
  float rstd = rsqrtf(ss * (1.0f / 128.0f) + 1e-5f);
  #pragma unroll
  for (int i = 0; i < 4; ++i) {
    int c = lane + 32 * i;
    float y = (v[i] - mu) * rstd * g[c] + b[c];
    y = 0.5f * y * (1.0f + erff(y * 0.70710678118654752f));
    x[c] = y;
    xb[c] = f2bf(y);
  }
}

// ---------------------------------------------------------------------------
// host-side helpers
// ---------------------------------------------------------------------------
static void launch_gemm(const ushort_t* A, int M, int K, int lda, const ushort_t* W,
                        const float* bias, float* C, int N, int ldc,
                        ushort_t* C2, int ldc2, hipStream_t s)
{
  dim3 grid((M + BM - 1) / BM, N / BN);
  gemm_bf16_kernel<<<grid, dim3(128), 0, s>>>(A, M, K, lda, W, bias, C, N, ldc, C2, ldc2);
}

static void launch_fill(float* p, long long n, float v, hipStream_t s) {
  fill_kernel<<<(int)((n + 255) / 256), 256, 0, s>>>(p, n, v);
}

static void launch_conv(const float* in, ushort_t* out, long long n, hipStream_t s) {
  convert_bf16_kernel<<<(int)((n + 255) / 256), 256, 0, s>>>(in, out, n);
}

extern "C" void kernel_launch(void* const* d_in, const int* in_sizes, int n_in,
                              void* d_out, int out_size, void* d_ws, size_t ws_size,
                              hipStream_t stream)
{
  (void)in_sizes; (void)n_in; (void)out_size; (void)ws_size;

  const float* x_paper  = (const float*)d_in[0];
  const float* x_author = (const float*)d_in[1];
  const int*   e_cites  = (const int*)d_in[2];
  const int*   e_writes = (const int*)d_in[3];
  const int*   e_rev    = (const int*)d_in[4];
  const float* in_w_p   = (const float*)d_in[5];
  const float* in_b_p   = (const float*)d_in[6];
  const float* out_w_p  = (const float*)d_in[7];
  const float* out_b_p  = (const float*)d_in[8];
  const float* in_w_a   = (const float*)d_in[9];
  const float* in_b_a   = (const float*)d_in[10];
  const float* out_w_a  = (const float*)d_in[11];
  const float* out_b_a  = (const float*)d_in[12];
  const float* kqv_w    = (const float*)d_in[13];
  const float* kqv_b    = (const float*)d_in[14];
  const float* aout_w   = (const float*)d_in[15];
  const float* aout_b   = (const float*)d_in[16];
  const float* skip     = (const float*)d_in[17];
  const float* a_rel    = (const float*)d_in[18];
  const float* m_rel    = (const float*)d_in[19];
  const float* p_rel    = (const float*)d_in[20];
  const float* ln_g     = (const float*)d_in[21];
  const float* ln_b     = (const float*)d_in[22];

  const int Np = 100000, Na = 20000;
  const int Ec = 320000, Ew = 160000, Er = 160000;
  const float NEG_INF = -__builtin_huge_valf();

  // bump allocator over workspace (16B-aligned chunks)
  float* ws = (float*)d_ws;
  size_t off = 0;
  auto alloc  = [&](size_t n) { float* p = ws + off; off += (n + 3) & ~(size_t)3; return p; };
  auto allocH = [&](size_t n) { return (ushort_t*)alloc((n + 1) / 2); };  // n ushorts

  // f32 buffers
  float* h_p    = alloc((size_t)Np * 128);
  float* h_a    = alloc((size_t)Na * 128);
  float* kqv_p  = alloc((size_t)Np * 384);
  float* kqv_a  = alloc((size_t)Na * 384);
  float* kt0    = alloc((size_t)Np * 128);
  float* vt0    = alloc((size_t)Np * 128);
  float* kt1    = alloc((size_t)Na * 128);
  float* vt1    = alloc((size_t)Na * 128);
  float* kt2    = alloc((size_t)Np * 128);
  float* vt2    = alloc((size_t)Np * 128);
  float* s0     = alloc((size_t)Ec * 4);
  float* s1     = alloc((size_t)Ew * 4);
  float* s2     = alloc((size_t)Er * 4);
  float* mx_p   = alloc((size_t)Np * 4);
  float* mx_a   = alloc((size_t)Na * 4);
  float* den_p  = alloc((size_t)Np * 4);
  float* den_a  = alloc((size_t)Na * 4);
  float* agg_p  = alloc((size_t)Np * 128);
  float* agg_a  = alloc((size_t)Na * 128);
  float* anew_p = alloc((size_t)Np * 128);
  float* anew_a = alloc((size_t)Na * 128);

  // bf16 buffers
  ushort_t* xp_bf    = allocH((size_t)Np * 128);
  ushort_t* xa_bf    = allocH((size_t)Na * 64);
  ushort_t* hp_bf    = allocH((size_t)Np * 128);
  ushort_t* ha_bf    = allocH((size_t)Na * 128);
  ushort_t* kqvp_bf  = allocH((size_t)Np * 384);
  ushort_t* kqva_bf  = allocH((size_t)Na * 384);
  ushort_t* aggp_bf  = allocH((size_t)Np * 128);
  ushort_t* agga_bf  = allocH((size_t)Na * 128);
  ushort_t* inwp_bf  = allocH(128 * 128);
  ushort_t* inwa_bf  = allocH(128 * 64);
  ushort_t* kqvw_bf  = allocH((size_t)2 * 2 * 3 * 128 * 128);
  ushort_t* aoutw_bf = allocH((size_t)2 * 2 * 128 * 128);
  ushort_t* outwp_bf = allocH(128 * 128);
  ushort_t* outwa_bf = allocH(128 * 128);
  ushort_t* bd_bf    = allocH(128 * 128);

  // one-time conversions: inputs + all weights to bf16
  launch_conv(x_paper,  xp_bf,  (long long)Np * 128, stream);
  launch_conv(x_author, xa_bf,  (long long)Na * 64,  stream);
  launch_conv(in_w_p,   inwp_bf,  128 * 128, stream);
  launch_conv(in_w_a,   inwa_bf,  128 * 64,  stream);
  launch_conv(kqv_w,    kqvw_bf,  (long long)2 * 2 * 3 * 128 * 128, stream);
  launch_conv(aout_w,   aoutw_bf, (long long)2 * 2 * 128 * 128, stream);
  launch_conv(out_w_p,  outwp_bf, 128 * 128, stream);
  launch_conv(out_w_a,  outwa_bf, 128 * 128, stream);

  // input projections: h = x @ in_w^T + in_b (dual-store bf16 for KQV GEMM)
  launch_gemm(xp_bf, Np, 128, 128, inwp_bf, in_b_p, h_p, 128, 128, hp_bf, 128, stream);
  launch_gemm(xa_bf, Na,  64,  64, inwa_bf, in_b_a, h_a, 128, 128, ha_bf, 128, stream);

  for (int l = 0; l < 2; ++l) {
    // fused KQV: [N,384] = h @ kqv_w[l][i]^T + kqv_b (dual-store bf16)
    launch_gemm(hp_bf, Np, 128, 128, kqvw_bf + (size_t)(l * 2 + 0) * 3 * 128 * 128,
                kqv_b + (size_t)(l * 2 + 0) * 384, kqv_p, 384, 384, kqvp_bf, 384, stream);
    launch_gemm(ha_bf, Na, 128, 128, kqvw_bf + (size_t)(l * 2 + 1) * 3 * 128 * 128,
                kqv_b + (size_t)(l * 2 + 1) * 384, kqv_a, 384, 384, kqva_bf, 384, stream);

    // per-relation K/V transforms (block-diagonal GEMMs over the K/V slices)
    struct RelSrc { const ushort_t* srcK; const ushort_t* srcV; int Ns; float* kt; float* vt; };
    RelSrc rel[3] = {
      { kqvp_bf, kqvp_bf + 256, Np, kt0, vt0 },   // paper  -> paper
      { kqva_bf, kqva_bf + 256, Na, kt1, vt1 },   // author -> paper
      { kqvp_bf, kqvp_bf + 256, Np, kt2, vt2 },   // paper  -> author
    };
    for (int j = 0; j < 3; ++j) {
      build_blockdiag_kernel<<<64, 256, 0, stream>>>(a_rel + (size_t)(l * 3 + j) * 4096, bd_bf);
      launch_gemm(rel[j].srcK, rel[j].Ns, 128, 384, bd_bf, nullptr, rel[j].kt, 128, 128,
                  nullptr, 0, stream);
      build_blockdiag_kernel<<<64, 256, 0, stream>>>(m_rel + (size_t)(l * 3 + j) * 4096, bd_bf);
      launch_gemm(rel[j].srcV, rel[j].Ns, 128, 384, bd_bf, nullptr, rel[j].vt, 128, 128,
                  nullptr, 0, stream);
    }

    // softmax stats + aggregation buffers
    launch_fill(mx_p,  (long long)Np * 4,   NEG_INF, stream);
    launch_fill(mx_a,  (long long)Na * 4,   NEG_INF, stream);
    launch_fill(den_p, (long long)Np * 4,   0.0f,    stream);
    launch_fill(den_a, (long long)Na * 4,   0.0f,    stream);
    launch_fill(agg_p, (long long)Np * 128, 0.0f,    stream);
    launch_fill(agg_a, (long long)Na * 128, 0.0f,    stream);

    // pass 1: scores + scatter max
    edge_score_kernel<<<(Ec + 7) / 8, 256, 0, stream>>>(e_cites,  Ec, kqv_p, kt0,
        p_rel + (size_t)(l * 3 + 0) * 4, s0, mx_p);
    edge_score_kernel<<<(Ew + 7) / 8, 256, 0, stream>>>(e_writes, Ew, kqv_p, kt1,
        p_rel + (size_t)(l * 3 + 1) * 4, s1, mx_p);
    edge_score_kernel<<<(Er + 7) / 8, 256, 0, stream>>>(e_rev,    Er, kqv_a, kt2,
        p_rel + (size_t)(l * 3 + 2) * 4, s2, mx_a);

    // pass 2: exp + denominator
    edge_expsum_kernel<<<(Ec * 4 + 255) / 256, 256, 0, stream>>>(e_cites,  Ec, s0, mx_p, den_p);
    edge_expsum_kernel<<<(Ew * 4 + 255) / 256, 256, 0, stream>>>(e_writes, Ew, s1, mx_p, den_p);
    edge_expsum_kernel<<<(Er * 4 + 255) / 256, 256, 0, stream>>>(e_rev,    Er, s2, mx_a, den_a);

    // pass 3: weighted message scatter
    edge_scatter_kernel<<<(int)(((long long)Ec * 128 + 255) / 256), 256, 0, stream>>>(
        e_cites,  Ec, s0, den_p, vt0, agg_p);
    edge_scatter_kernel<<<(int)(((long long)Ew * 128 + 255) / 256), 256, 0, stream>>>(
        e_writes, Ew, s1, den_p, vt1, agg_p);
    edge_scatter_kernel<<<(int)(((long long)Er * 128 + 255) / 256), 256, 0, stream>>>(
        e_rev,    Er, s2, den_a, vt2, agg_a);

    // gelu(agg) -> bf16, then anew = gelu(agg) @ aout_w^T + aout_b
    gelu_bf_kernel<<<(int)(((long long)Np * 128 + 255) / 256), 256, 0, stream>>>(
        agg_p, aggp_bf, (long long)Np * 128);
    gelu_bf_kernel<<<(int)(((long long)Na * 128 + 255) / 256), 256, 0, stream>>>(
        agg_a, agga_bf, (long long)Na * 128);
    launch_gemm(aggp_bf, Np, 128, 128, aoutw_bf + (size_t)(l * 2 + 0) * 16384,
                aout_b + (size_t)(l * 2 + 0) * 128, anew_p, 128, 128, nullptr, 0, stream);
    launch_gemm(agga_bf, Na, 128, 128, aoutw_bf + (size_t)(l * 2 + 1) * 16384,
                aout_b + (size_t)(l * 2 + 1) * 128, anew_a, 128, 128, nullptr, 0, stream);

    // skip blend (f32)
    blend_kernel<<<(int)(((long long)Np * 128 + 255) / 256), 256, 0, stream>>>(
        h_p, anew_p, (long long)Np * 128, skip, l * 2 + 0);
    blend_kernel<<<(int)(((long long)Na * 128 + 255) / 256), 256, 0, stream>>>(
        h_a, anew_a, (long long)Na * 128, skip, l * 2 + 1);

    // LayerNorm + GELU (writes f32 h + bf16 h for the next GEMM consumer)
    ln_gelu_kernel<<<(Np + 7) / 8, 256, 0, stream>>>(h_p, hp_bf, Np,
        ln_g + (size_t)(l * 2 + 0) * 128, ln_b + (size_t)(l * 2 + 0) * 128);
    ln_gelu_kernel<<<(Na + 7) / 8, 256, 0, stream>>>(h_a, ha_bf, Na,
        ln_g + (size_t)(l * 2 + 1) * 128, ln_b + (size_t)(l * 2 + 1) * 128);
  }

  // output heads
  float* out = (float*)d_out;
  launch_gemm(hp_bf, Np, 128, 128, outwp_bf, out_b_p, out, 128, 128, nullptr, 0, stream);
  launch_gemm(ha_bf, Na, 128, 128, outwa_bf, out_b_a, out + (size_t)Np * 128, 128, 128,
              nullptr, 0, stream);
}
